// S4Ternary_44212393345427
// MI455X (gfx1250) — 8-bit matrix operand order repaired and checked by reading, compiled, not run on hardware
//
#include <hip/hip_runtime.h>
#include <stdint.h>

typedef int v8i __attribute__((ext_vector_type(8)));

#define QDIM 2688
#define DM 256
#define ROWS_TOTAL (8*16384)
#define TILES (ROWS_TOTAL/16)
#define WAVES 4
#define NSUM 384

#define WS_SCAL    0
#define WS_QFLAT   8
#define WS_SUMMARY 264
#define WS_BPACK   520

__device__ __forceinline__ int clampi(int v, int lo, int hi){ return v<lo?lo:(v>hi?hi:v); }

__device__ float block_reduce_sum(float v, float* red){
  int tid = threadIdx.x;
  red[tid] = v; __syncthreads();
  for (int s = blockDim.x>>1; s > 0; s >>= 1){ if (tid < s) red[tid] += red[tid+s]; __syncthreads(); }
  float r = red[0]; __syncthreads(); return r;
}
__device__ float block_reduce_max(float v, float* red){
  int tid = threadIdx.x;
  red[tid] = v; __syncthreads();
  for (int s = blockDim.x>>1; s > 0; s >>= 1){ if (tid < s) red[tid] = fmaxf(red[tid], red[tid+s]); __syncthreads(); }
  float r = red[0]; __syncthreads(); return r;
}

__global__ __launch_bounds__(256) void scales_kernel(
    const float* __restrict__ banks, const float* __restrict__ wq,
    const float* __restrict__ wk, const float* __restrict__ wv,
    const float* __restrict__ wsum, float* __restrict__ scal){
  __shared__ float red[256];
  int tid = threadIdx.x;
  float s = 0.f;
  for (int i = tid; i < DM*QDIM; i += 256) s += fabsf(wq[i]);
  s = block_reduce_sum(s, red);
  if (tid == 0) scal[0] = 1.f / fmaxf(s / (float)(DM*QDIM), 1e-5f);
  s = 0.f;
  for (int i = tid; i < DM*DM; i += 256) s += fabsf(wk[i]);
  s = block_reduce_sum(s, red);
  if (tid == 0) scal[1] = 1.f / fmaxf(s / (float)(DM*DM), 1e-5f);
  s = 0.f;
  for (int i = tid; i < DM*DM; i += 256) s += fabsf(wv[i]);
  s = block_reduce_sum(s, red);
  if (tid == 0) scal[2] = 1.f / fmaxf(s / (float)(DM*DM), 1e-5f);
  s = 0.f;
  for (int i = tid; i < NSUM*DM; i += 256) s += fabsf(wsum[i]);
  s = block_reduce_sum(s, red);
  if (tid == 0) scal[3] = 1.f / fmaxf(s / (float)(NSUM*DM), 1e-5f);
  float m = 0.f;
  for (int i = tid; i < QDIM; i += 256) m = fmaxf(m, fabsf(banks[i]));
  m = block_reduce_max(m, red);
  if (tid == 0) scal[4] = 127.f / fmaxf(m, 1e-5f);
}

__global__ __launch_bounds__(256) void qflat_kernel(
    const float* __restrict__ banks, const float* __restrict__ wq,
    const float* __restrict__ scal, float* __restrict__ qflat){
  int t = threadIdx.x;
  float saq = scal[4], swq = scal[0];
  int acc = 0;
  for (int k = 0; k < QDIM; ++k){
    int qa = clampi((int)rintf(banks[k]*saq), -128, 127);
    int qw = clampi((int)rintf(wq[t*QDIM + k]*swq), -1, 1);
    acc += qa*qw;
  }
  qflat[t] = (float)acc / (saq*swq);
}

__global__ __launch_bounds__(256) void pack_kernel(
    const float* __restrict__ wk, const float* __restrict__ wv,
    const float* __restrict__ scal, uint32_t* __restrict__ bpack){
  int idx = blockIdx.x*256 + threadIdx.x;
  int mat = idx >> 14;
  int r   = idx & 16383;
  int ntile = r >> 10;
  int kb    = (r >> 8) & 3;
  int ln    = (r >> 3) & 31;
  int j     = r & 7;
  int n = ntile*16 + (ln & 15);
  int kbase = kb*64 + ((j>>2)<<5) + ((ln>>4)<<4) + ((j&3)<<2);
  const float* W = mat ? wv : wk;
  float sw = mat ? scal[2] : scal[1];
  uint32_t dw = 0;
  #pragma unroll
  for (int b2 = 0; b2 < 4; ++b2){
    int q = clampi((int)rintf(W[n*DM + kbase + b2]*sw), -1, 1);
    dw |= ((uint32_t)(uint8_t)(int8_t)q) << (8*b2);
  }
  bpack[idx] = dw;
}

__global__ void init_kernel(float* __restrict__ summary){ summary[threadIdx.x] = 0.f; }

__global__ __launch_bounds__(WAVES*32) void fused_rows_kernel(
    const float* __restrict__ residual, const float* __restrict__ norm_w,
    const uint32_t* __restrict__ bpack, const float* __restrict__ qflat,
    const float* __restrict__ aw, const float* __restrict__ scal,
    float* __restrict__ logits, float* __restrict__ summary, int mode){
  extern __shared__ uint4 smem_v4[];
  uint32_t* Blds  = (uint32_t*)smem_v4;
  uint32_t* Alds  = Blds + 16384;
  float*    rowsc = (float*)(Alds + WAVES*1024);
  float*    awcs  = rowsc + WAVES*16;
  float*    qlds  = awcs + WAVES*16;
  float*    sumlds= qlds + 256;

  const int tid = threadIdx.x, lane = tid & 31, wave = tid >> 5;
  for (int i = tid; i < 16384; i += blockDim.x) Blds[i] = bpack[i];
  for (int i = tid; i < 256; i += blockDim.x){ qlds[i] = qflat[i]; sumlds[i] = 0.f; }
  __syncthreads();

  const int tile = blockIdx.x*WAVES + wave;
  const int m = lane & 15, half = lane >> 4;
  const int row = tile*16 + m;
  const float* px = residual + (size_t)row*DM + half*128;
  const float* pg = norm_w + half*128;

  float ss = 0.f, mx = 0.f;
  #pragma unroll 8
  for (int i = 0; i < 128; i += 4){
    float4 x = *(const float4*)(px + i);
    float4 g = *(const float4*)(pg + i);
    ss += x.x*x.x + x.y*x.y + x.z*x.z + x.w*x.w;
    float a0=fabsf(x.x*g.x), a1=fabsf(x.y*g.y), a2=fabsf(x.z*g.z), a3=fabsf(x.w*g.w);
    mx = fmaxf(mx, fmaxf(fmaxf(a0,a1), fmaxf(a2,a3)));
  }
  ss += __shfl_xor(ss, 16, 32);
  mx  = fmaxf(mx, __shfl_xor(mx, 16, 32));
  const float rs   = rsqrtf(ss*(1.f/256.f) + 1e-6f);
  const float sact = 127.f / fmaxf(rs*mx, 1e-5f);
  const float qsc  = rs*sact;

  uint32_t* arow = Alds + wave*1024 + m*64 + half*32;
  #pragma unroll 8
  for (int i = 0; i < 128; i += 4){
    float4 x = *(const float4*)(px + i);
    float4 g = *(const float4*)(pg + i);
    int q0 = clampi((int)rintf(x.x*g.x*qsc), -128, 127);
    int q1 = clampi((int)rintf(x.y*g.y*qsc), -128, 127);
    int q2 = clampi((int)rintf(x.z*g.z*qsc), -128, 127);
    int q3 = clampi((int)rintf(x.w*g.w*qsc), -128, 127);
    arow[i>>2] = (uint32_t)(uint8_t)q0 | ((uint32_t)(uint8_t)q1<<8) |
                 ((uint32_t)(uint8_t)q2<<16) | ((uint32_t)(uint8_t)q3<<24);
  }
  const float sw   = (mode == 0) ? scal[1] : scal[2];
  const float crow = 1.f / (sact*sw);
  if (half == 0){
    rowsc[wave*16 + m] = crow;
    if (mode == 1) awcs[wave*16 + m] = aw[row]*crow*0.125f;
  }
  __syncthreads();

  uint32_t af[4][8];
  #pragma unroll
  for (int kb = 0; kb < 4; ++kb){
    #pragma unroll
    for (int j = 0; j < 8; ++j){
      int off = ((j>>2)<<5) + (half<<4) + ((j&3)<<2);
      af[kb][j] = Alds[wave*1024 + m*64 + (kb<<4) + (off>>2)];
    }
  }

  float attnp[8] = {0,0,0,0,0,0,0,0};
  for (int nt = 0; nt < 16; ++nt){
    v8i acc = {};
    #pragma unroll
    for (int kb = 0; kb < 4; ++kb){
      v8i a, b;
      #pragma unroll
      for (int j = 0; j < 8; ++j) a[j] = (int)af[kb][j];
      const uint4* bq = (const uint4*)(Blds + (size_t)(((nt<<2)+kb)*32 + lane)*8);
      uint4 b0 = bq[0], b1 = bq[1];
      b[0]=(int)b0.x; b[1]=(int)b0.y; b[2]=(int)b0.z; b[3]=(int)b0.w;
      b[4]=(int)b1.x; b[5]=(int)b1.y; b[6]=(int)b1.z; b[7]=(int)b1.w;
      acc = __builtin_amdgcn_wmma_i32_16x16x64_iu8(true, a, true, b, acc, false, false);
    }
    if (mode == 0){
      const float qv = qlds[(nt<<4) + m];
      #pragma unroll
      for (int r = 0; r < 8; ++r) attnp[r] += qv*(float)acc[r];
    } else {
      float sv = 0.f;
      #pragma unroll
      for (int r = 0; r < 8; ++r) sv += awcs[wave*16 + r + (half<<3)]*(float)acc[r];
      atomicAdd(&sumlds[(nt<<4) + m], sv);
    }
  }

  if (mode == 0){
    #pragma unroll
    for (int r = 0; r < 8; ++r){
      float v = attnp[r];
      v += __shfl_xor(v, 8, 16);
      v += __shfl_xor(v, 4, 16);
      v += __shfl_xor(v, 2, 16);
      v += __shfl_xor(v, 1, 16);
      attnp[r] = v;
    }
    if ((lane & 15) == 0){
      #pragma unroll
      for (int r = 0; r < 8; ++r){
        int mr = r + (half<<3);
        logits[tile*16 + mr] = attnp[r]*rowsc[wave*16 + mr]*0.0625f;
      }
    }
  } else {
    __syncthreads();
    for (int i = tid; i < 256; i += blockDim.x) atomicAdd(&summary[i], sumlds[i]);
  }
}

__global__ __launch_bounds__(256) void softmax_kernel(float* __restrict__ W){
  __shared__ float red[256];
  const int b = blockIdx.x, tid = threadIdx.x;
  float* L = W + (size_t)b*16384;
  float mx = -3.4e38f;
  for (int i = tid; i < 16384; i += 256) mx = fmaxf(mx, L[i]);
  mx = block_reduce_max(mx, red);
  float s = 0.f;
  for (int i = tid; i < 16384; i += 256) s += expf(L[i] - mx);
  s = block_reduce_sum(s, red);
  float inv = 1.f / s;
  for (int i = tid; i < 16384; i += 256) L[i] = expf(L[i] - mx)*inv;
}

__global__ __launch_bounds__(256) void final_kernel(
    const float* __restrict__ summary, const float* __restrict__ wsum,
    const float* __restrict__ scal, float* __restrict__ out){
  __shared__ float red[256];
  __shared__ int qsi[256];
  int tid = threadIdx.x;
  float mxa = block_reduce_max(fabsf(summary[tid]), red);
  float sas = 127.f / fmaxf(mxa, 1e-5f);
  qsi[tid] = clampi((int)rintf(summary[tid]*sas), -128, 127);
  __syncthreads();
  float sws = scal[3];
  for (int o = tid; o < NSUM; o += 256){
    int acc = 0;
    for (int k = 0; k < DM; ++k){
      int qw = clampi((int)rintf(wsum[o*DM + k]*sws), -1, 1);
      acc += qsi[k]*qw;
    }
    out[o] = (float)acc / (sas*sws);
  }
}

extern "C" void kernel_launch(void* const* d_in, const int* in_sizes, int n_in,
                              void* d_out, int out_size, void* d_ws, size_t ws_size,
                              hipStream_t stream){
  (void)in_sizes; (void)n_in; (void)out_size; (void)ws_size;
  const float* banks    = (const float*)d_in[0];
  const float* residual = (const float*)d_in[1];
  const float* wq       = (const float*)d_in[2];
  const float* wk       = (const float*)d_in[3];
  const float* wv       = (const float*)d_in[4];
  const float* wsum     = (const float*)d_in[5];
  const float* norm_w   = (const float*)d_in[6];
  float* out = (float*)d_out;
  float* ws  = (float*)d_ws;

  float*    scal    = ws + WS_SCAL;
  float*    qflat   = ws + WS_QFLAT;
  float*    summary = ws + WS_SUMMARY;
  uint32_t* bpack   = (uint32_t*)(ws + WS_BPACK);
  float*    awout   = out + NSUM;

  const size_t smem = 16384*4 + WAVES*1024*4 + WAVES*16*4*2 + 256*4*2;
  hipFuncSetAttribute(reinterpret_cast<const void*>(fused_rows_kernel),
                      hipFuncAttributeMaxDynamicSharedMemorySize, (int)smem);

  scales_kernel<<<1, 256, 0, stream>>>(banks, wq, wk, wv, wsum, scal);
  qflat_kernel<<<1, 256, 0, stream>>>(banks, wq, scal, qflat);
  pack_kernel<<<128, 256, 0, stream>>>(wk, wv, scal, bpack);
  init_kernel<<<1, 256, 0, stream>>>(summary);
  fused_rows_kernel<<<TILES/WAVES, WAVES*32, smem, stream>>>(
      residual, norm_w, bpack, qflat, awout, scal, awout, summary, 0);
  softmax_kernel<<<8, 256, 0, stream>>>(awout);
  fused_rows_kernel<<<TILES/WAVES, WAVES*32, smem, stream>>>(
      residual, norm_w, bpack + 16384, qflat, awout, scal, awout, summary, 1);
  final_kernel<<<1, 256, 0, stream>>>(summary, wsum, scal, out);
}
